// ConvNeXtBlock_RNN_69999376990712
// MI455X (gfx1250) — matrix operand order repaired and checked by reading, not compiled, not run on hardware
//
#include <hip/hip_runtime.h>
#include <math.h>

#define DIMC 512
#define IDIMC 1024
#define BB 8
#define TT 1024

typedef __bf16 bf16;
typedef bf16 v16bf __attribute__((ext_vector_type(16)));
typedef bf16 v8bf  __attribute__((ext_vector_type(8)));
typedef float v8f  __attribute__((ext_vector_type(8)));
typedef int   v4i  __attribute__((ext_vector_type(4)));

static __device__ inline v16bf pack16(v8bf lo, v8bf hi) {
    union { v16bf v; v8bf h[2]; } u;
    u.h[0] = lo; u.h[1] = hi;
    return u.v;
}

static __device__ inline v8f wmma_bf16(v16bf a, v16bf b, v8f c) {
    // v_wmma_f32_16x16x32_bf16: D = A(16x32) * B(32x16) + C
    return __builtin_amdgcn_wmma_f32_16x16x32_bf16(false, a, false, b, (short)0, c, false, false);
}

static __device__ inline float gelu_exact(float x) {
    return 0.5f * x * (1.0f + erff(x * 0.70710678118654752f));
}

// ---------------------------------------------------------------------------
// gfx1250 async global->LDS copy (ASYNCcnt path) with inline-asm fallback
// ---------------------------------------------------------------------------
static __device__ inline void async_load_b128(const bf16* gsrc, bf16* ldst) {
#if __has_builtin(__builtin_amdgcn_global_load_async_to_lds_b128)
    __builtin_amdgcn_global_load_async_to_lds_b128(
        (__attribute__((address_space(1))) v4i*)(void*)gsrc,
        (__attribute__((address_space(3))) v4i*)(void*)ldst, 0, 0);
#else
    unsigned loff = (unsigned)(size_t)(__attribute__((address_space(3))) char*)(void*)ldst;
    asm volatile("global_load_async_to_lds_b128 %0, %1, off"
                 :: "v"(loff), "v"(gsrc) : "memory");
#endif
}

#if __has_builtin(__builtin_amdgcn_s_wait_asynccnt)
#define WAIT_ASYNC(n) __builtin_amdgcn_s_wait_asynccnt(n)
#else
#define WAIT_ASYNC(n) asm volatile("s_wait_asynccnt %0" :: "i"(n) : "memory")
#endif

// ---------------------------------------------------------------------------
// fp32 -> bf16 weight conversion
// ---------------------------------------------------------------------------
__global__ void f32_to_bf16_k(const float* __restrict__ src, bf16* __restrict__ dst, int n) {
    int i = blockIdx.x * blockDim.x + threadIdx.x;
    if (i < n) dst[i] = (bf16)src[i];
}

// ---------------------------------------------------------------------------
// depthwise conv (7 taps, pad 3) + bias + exact GELU; x:[B,D,T] -> hb:[B*T, D] bf16
// ---------------------------------------------------------------------------
__global__ void dwconv0_gelu_k(const float* __restrict__ x, const float* __restrict__ w,
                               const float* __restrict__ b, bf16* __restrict__ hb) {
    int t = blockIdx.x * blockDim.x + threadIdx.x;
    int d = blockIdx.y, bb = blockIdx.z;
    const float* xp = x + ((size_t)bb * DIMC + d) * TT;
    float s = b[d];
#pragma unroll
    for (int k = 0; k < 7; ++k) {
        int tt = t + k - 3;
        if (tt >= 0 && tt < TT) s += w[d * 7 + k] * xp[tt];
    }
    hb[((size_t)bb * TT + t) * DIMC + d] = (bf16)gelu_exact(s);
}

// final depthwise conv + bias + residual; y2:[B*T, D] bf16 -> out:[B,D,T] fp32
__global__ void dwconv1_res_k(const float* __restrict__ x, const bf16* __restrict__ y2,
                              const float* __restrict__ w, const float* __restrict__ b,
                              float* __restrict__ out) {
    int t = blockIdx.x * blockDim.x + threadIdx.x;
    int d = blockIdx.y, bb = blockIdx.z;
    float s = b[d];
#pragma unroll
    for (int k = 0; k < 7; ++k) {
        int tt = t + k - 3;
        if (tt >= 0 && tt < TT) s += w[d * 7 + k] * (float)y2[((size_t)bb * TT + tt) * DIMC + d];
    }
    size_t idx = ((size_t)bb * DIMC + d) * TT + t;
    out[idx] = x[idx] + s;
}

// ---------------------------------------------------------------------------
// WMMA GEMM: out[M,N] = act(A[M,K] @ W[N,K]^T + bias)
// 8 waves/block share one 16-row M tile; A is staged through double-buffered
// LDS with global_load_async_to_lds_b128 (one b128 per thread per 128-K chunk,
// prefetch overlapped with WMMA, s_wait_asynccnt pipelining). W streams on the
// direct global_load_b128 path (per-wave unique columns, L2-served).
// A-frag (16x32): lane(lo) row=m holds K {k..k+15}; lane(hi) K {k+16..k+31}.
// B-frag (32x16): lane(lo) col=n holds K {k..k+15}; lane(hi) K {k+16..k+31}.
// ---------------------------------------------------------------------------
#define AK   128                // K elements staged per chunk
#define APAD 136                // padded LDS row stride (272B): conflict-free b128

__global__ void gemm_wmma_k(const bf16* __restrict__ A, const bf16* __restrict__ W,
                            const float* __restrict__ bias,
                            float* __restrict__ outF, bf16* __restrict__ outB,
                            int M, int N, int K, int fuseGelu) {
    __shared__ __align__(16) bf16 abuf[2][16 * APAD];

    const int tid  = threadIdx.x;
    const int lane = tid & 31;
    const int wave = tid >> 5;
    const int ntn  = N >> 4;
    const int tile = blockIdx.x * 8 + wave;
    const int tm   = tile / ntn;           // uniform across block (ntn % 8 == 0)
    const int tn   = tile % ntn;
    const int hi   = lane >> 4;
    const int lr   = lane & 15;
    const int arow0 = tm << 4;
    const int bcol  = (tn << 4) + lr;

    // this thread's slot in the A staging pattern: row r, 8-elem chunk c
    const int sr = tid >> 4;
    const int sc = (tid & 15) << 3;
    const bf16* Ag = A + (size_t)(arow0 + sr) * K + sc;

    const bf16* Wp = W + (size_t)bcol * K + (hi ? 16 : 0);
    const bf16* Alds = (const bf16*)0;

    const int nk = K / AK;
    // prefetch chunk 0
    async_load_b128(Ag, &abuf[0][sr * APAD + sc]);

    v8f acc = {};
    for (int kc = 0; kc < nk; ++kc) {
        if (kc + 1 < nk) {
            async_load_b128(Ag + (size_t)(kc + 1) * AK, &abuf[(kc + 1) & 1][sr * APAD + sc]);
            WAIT_ASYNC(1);   // chunk kc complete (in-order), kc+1 in flight
        } else {
            WAIT_ASYNC(0);
        }
        __syncthreads();

        Alds = &abuf[kc & 1][(size_t)lr * APAD + (hi ? 16 : 0)];
        const bf16* Wk = Wp + (size_t)kc * AK;
#pragma unroll
        for (int ks = 0; ks < AK / 32; ++ks) {
            v16bf af = pack16(*(const v8bf*)(Alds + ks * 32),
                              *(const v8bf*)(Alds + ks * 32 + 8));
            v16bf wf = pack16(*(const v8bf*)(Wk + ks * 32),
                              *(const v8bf*)(Wk + ks * 32 + 8));
            acc = wmma_bf16(af, wf, acc);
        }
        __syncthreads();
    }

    const float bv = bias ? bias[bcol] : 0.0f;
#pragma unroll
    for (int r = 0; r < 8; ++r) {
        int row = arow0 + r + (hi << 3);   // C layout: vgpr r -> M = r (+8 for hi lanes)
        float v = acc[r] + bv;
        if (fuseGelu) v = gelu_exact(v);
        size_t idx = (size_t)row * N + bcol;
        if (outF) outF[idx] = v;
        if (outB) outB[idx] = (bf16)v;
    }
}

// ---------------------------------------------------------------------------
// grid-wide split barrier (persistent scan kernel)
// ---------------------------------------------------------------------------
static __device__ inline void grid_sync(int* bar, int nblk, int gen) {
    __syncthreads();
    if (threadIdx.x == 0) {
        __threadfence();
        if (atomicAdd(&bar[0], 1) == nblk - 1) {
            bar[0] = 0;
            __threadfence();
            __hip_atomic_store(&bar[1], gen, __ATOMIC_RELEASE, __HIP_MEMORY_SCOPE_AGENT);
        } else {
            while (__hip_atomic_load(&bar[1], __ATOMIC_ACQUIRE, __HIP_MEMORY_SCOPE_AGENT) < gen) {
                __builtin_amdgcn_s_sleep(1);
            }
        }
    }
    __syncthreads();
}

// ---------------------------------------------------------------------------
// Persistent RNN scan: h_t = tanh(U[t] + h_{t-1} @ W^T + bias)
// 16 blocks x 4 waves = 64 waves; each wave owns 16 output columns.
// Recurrent weight slice (64 cols x 1024 K, bf16) resident in 132 KB LDS
// (async-staged once), padded stride for conflict-free ds_load_b128.
// Hidden state (8 x 1024 bf16) double-buffered in global (L2-resident);
// one grid barrier per timestep.
// ---------------------------------------------------------------------------
#define SC_BLOCKS 16
#define SC_WAVES  4
#define SC_PADK   (IDIMC + 8)   // 1032 elements -> 2064B row stride, 16B aligned

__global__ void rnn_scan_k(const float* __restrict__ U, const bf16* __restrict__ W,
                           const float* __restrict__ bias, const float* __restrict__ h_init,
                           bf16* __restrict__ Hout, bf16* __restrict__ hbuf0,
                           bf16* __restrict__ hbuf1, int* __restrict__ bar) {
    __shared__ __align__(16) bf16 smem[SC_WAVES * 16 * SC_PADK];

    const int tid  = threadIdx.x;
    const int lane = tid & 31;
    const int wave = tid >> 5;
    const int hi   = lane >> 4;
    const int lr   = lane & 15;
    const int n0b  = blockIdx.x * (SC_WAVES * 16);

    // async-stage this block's 64x1024 bf16 weight slice into LDS (padded rows)
    for (int i = tid; i < SC_WAVES * 16 * (IDIMC / 8); i += blockDim.x) {
        int r = i >> 7;            // 128 8-elem chunks per row
        int c = (i & 127) << 3;
        async_load_b128(&W[(size_t)(n0b + r) * IDIMC + c], &smem[(size_t)r * SC_PADK + c]);
    }
    WAIT_ASYNC(0);

    const int col = n0b + (wave << 4) + lr;   // this lane's global output column
    const float bv = bias[col];

    // init hidden-state buffer 0 with broadcast starter row
    if (!hi) {
        bf16 s = (bf16)h_init[col];
        for (int b = 0; b < BB; ++b) hbuf0[b * IDIMC + col] = s;
    }
    __syncthreads();

    int gen = 1;
    grid_sync(bar, SC_BLOCKS, gen++);

    const bf16* sWp = &smem[(size_t)((wave << 4) + lr) * SC_PADK + (hi ? 16 : 0)];
    const bool arow_ok = (lr < BB);

    for (int t = 0; t < TT; ++t) {
        const bf16* hprev = (t & 1) ? hbuf1 : hbuf0;
        bf16*       hnext = (t & 1) ? hbuf0 : hbuf1;

        const bf16* Ap = hprev + (size_t)lr * IDIMC + (hi ? 16 : 0);
        v8f acc = {};
        for (int k0 = 0; k0 < IDIMC; k0 += 32) {
            v8bf alo = {}, ahi = {};
            if (arow_ok) {
                alo = *(const v8bf*)(Ap + k0);
                ahi = *(const v8bf*)(Ap + k0 + 8);
            }
            v16bf af = pack16(alo, ahi);
            v16bf wf = pack16(*(const v8bf*)(sWp + k0), *(const v8bf*)(sWp + k0 + 8));
            acc = wmma_bf16(af, wf, acc);
        }

        if (!hi) {  // lo lanes hold C rows 0..7 == all 8 batches
#pragma unroll
            for (int r = 0; r < BB; ++r) {
                float v = acc[r] + U[((size_t)r * TT + t) * IDIMC + col] + bv;
                v = tanhf(v);
                bf16 vb = (bf16)v;
                hnext[r * IDIMC + col] = vb;
                Hout[((size_t)r * TT + t) * IDIMC + col] = vb;
            }
        }
        grid_sync(bar, SC_BLOCKS, gen++);
    }
}

// ---------------------------------------------------------------------------
extern "C" void kernel_launch(void* const* d_in, const int* in_sizes, int n_in,
                              void* d_out, int out_size, void* d_ws, size_t ws_size,
                              hipStream_t stream) {
    (void)in_sizes; (void)n_in; (void)out_size; (void)ws_size;

    const float* x      = (const float*)d_in[0];
    const float* w_dw0  = (const float*)d_in[1];
    const float* b_dw0  = (const float*)d_in[2];
    const float* w_dw1  = (const float*)d_in[3];
    const float* b_dw1  = (const float*)d_in[4];
    const float* w_join = (const float*)d_in[5];
    const float* b_join = (const float*)d_in[6];
    const float* w_ih0  = (const float*)d_in[7];
    const float* w_hh0  = (const float*)d_in[8];
    const float* b_ih0  = (const float*)d_in[9];
    const float* b_hh0  = (const float*)d_in[10];
    const float* w_ih1  = (const float*)d_in[11];
    const float* w_hh1  = (const float*)d_in[12];
    const float* b_ih1  = (const float*)d_in[13];
    const float* b_hh1  = (const float*)d_in[14];
    const float* w_u1   = (const float*)d_in[15];
    const float* b_u1   = (const float*)d_in[16];
    const float* w_u2   = (const float*)d_in[17];
    const float* b_u2   = (const float*)d_in[18];
    const float* starter= (const float*)d_in[19];
    float* out = (float*)d_out;
    char*  ws  = (char*)d_ws;

    const size_t MB = 1024ull * 1024ull;
    // bf16 weights
    bf16* wjB   = (bf16*)(ws + 0 * MB);     // 1 MB
    bf16* wih0B = (bf16*)(ws + 1 * MB);     // 2 MB
    bf16* whh0B = (bf16*)(ws + 3 * MB);     // 2 MB
    bf16* wih1B = (bf16*)(ws + 5 * MB);     // 2 MB
    bf16* whh1B = (bf16*)(ws + 7 * MB);     // 2 MB
    bf16* wu1B  = (bf16*)(ws + 9 * MB);     // 2 MB
    bf16* wu2B  = (bf16*)(ws + 11 * MB);    // 1 MB
    bf16* hcA   = (bf16*)(ws + 12 * MB);            // 16 KB
    bf16* hcB   = (bf16*)(ws + 12 * MB + 64 * 1024);// 16 KB
    int*  bars  = (int*) (ws + 12 * MB + 128 * 1024);
    char* S1 = ws + 16 * MB;   // 32 MB slot
    char* S2 = ws + 48 * MB;   // 32 MB slot

    // aliased activation buffers (sequential lifetimes)
    bf16*  hb     = (bf16*)S1;   // [8192, 512]
    bf16*  a_join = (bf16*)S2;   // [8192, 1024]
    float* u0     = (float*)S1;  // [8192, 1024]
    bf16*  H0     = (bf16*)S2;   // [8192, 1024]
    float* v      = (float*)S1;  // [8192, 1024]
    bf16*  ys     = (bf16*)S2;   // [8192, 1024]
    bf16*  y1     = (bf16*)S1;   // [8192, 1024]
    bf16*  y2     = (bf16*)S2;   // [8192, 512]

    (void)hipMemsetAsync(bars, 0, 64, stream);

    // weight conversion to bf16
    f32_to_bf16_k<<<(IDIMC * DIMC) / 256, 256, 0, stream>>>(w_join, wjB, IDIMC * DIMC);
    f32_to_bf16_k<<<(IDIMC * IDIMC) / 256, 256, 0, stream>>>(w_ih0, wih0B, IDIMC * IDIMC);
    f32_to_bf16_k<<<(IDIMC * IDIMC) / 256, 256, 0, stream>>>(w_hh0, whh0B, IDIMC * IDIMC);
    f32_to_bf16_k<<<(IDIMC * IDIMC) / 256, 256, 0, stream>>>(w_ih1, wih1B, IDIMC * IDIMC);
    f32_to_bf16_k<<<(IDIMC * IDIMC) / 256, 256, 0, stream>>>(w_hh1, whh1B, IDIMC * IDIMC);
    f32_to_bf16_k<<<(IDIMC * IDIMC) / 256, 256, 0, stream>>>(w_u1, wu1B, IDIMC * IDIMC);
    f32_to_bf16_k<<<(DIMC * IDIMC) / 256, 256, 0, stream>>>(w_u2, wu2B, DIMC * IDIMC);

    dim3 cgrid(TT / 256, DIMC, BB);
    dwconv0_gelu_k<<<cgrid, 256, 0, stream>>>(x, w_dw0, b_dw0, hb);

    const int M = BB * TT;  // 8192
    // join: [8192,512] @ w_join[1024,512]^T -> bf16
    gemm_wmma_k<<<(M / 16) * (IDIMC / 16) / 8, 256, 0, stream>>>(
        hb, wjB, b_join, nullptr, a_join, M, IDIMC, DIMC, 0);
    // u0 = a_join @ w_ih0^T + b_ih0 -> fp32
    gemm_wmma_k<<<(M / 16) * (IDIMC / 16) / 8, 256, 0, stream>>>(
        a_join, wih0B, b_ih0, u0, nullptr, M, IDIMC, IDIMC, 0);

    // layer-0 recurrence
    rnn_scan_k<<<SC_BLOCKS, SC_WAVES * 32, 0, stream>>>(
        u0, whh0B, b_hh0, starter + 0, H0, hcA, hcB, bars);

    // v = H0 @ w_ih1^T + b_ih1 -> fp32
    gemm_wmma_k<<<(M / 16) * (IDIMC / 16) / 8, 256, 0, stream>>>(
        H0, wih1B, b_ih1, v, nullptr, M, IDIMC, IDIMC, 0);

    // layer-1 recurrence
    rnn_scan_k<<<SC_BLOCKS, SC_WAVES * 32, 0, stream>>>(
        v, whh1B, b_hh1, starter + IDIMC, ys, hcA, hcB, bars + 8);

    // y1 = gelu(ys @ w_u1^T + b_u1) -> bf16
    gemm_wmma_k<<<(M / 16) * (IDIMC / 16) / 8, 256, 0, stream>>>(
        ys, wu1B, b_u1, nullptr, y1, M, IDIMC, IDIMC, 1);
    // y2 = gelu(y1 @ w_u2^T + b_u2) -> bf16 [8192, 512]
    gemm_wmma_k<<<(M / 16) * (DIMC / 16) / 8, 256, 0, stream>>>(
        y1, wu2B, b_u2, nullptr, y2, M, DIMC, IDIMC, 1);

    dwconv1_res_k<<<cgrid, 256, 0, stream>>>(x, y2, w_dw1, b_dw1, out);
}
